// ScaleDotAttentionLayer_82386062672072
// MI455X (gfx1250) — matrix operand order repaired and checked by reading, not compiled, not run on hardware
//
#include <hip/hip_runtime.h>
#include <hip/hip_bf16.h>

typedef __attribute__((ext_vector_type(16))) _Float16 v16h;
typedef __attribute__((ext_vector_type(8)))  _Float16 v8h;
typedef __attribute__((ext_vector_type(8)))  float    v8f;

#define ATT_B 4
#define ATT_H 8
#define ATT_S 2048
#define ATT_D 64
// Work in log2 domain: fold (1/sqrt(64)) * log2(e) into Q.
#define SCALE_L2E 0.18033686011112042f
// Masked score in log2 domain: -1e9 * log2(e)
#define NEG_L2E  (-1.4426950408889634e9f)
#define KSTRIDE 72         // padded LDS row stride in halfs (144B = 36 dwords)

// Build a v16h from two 8-half (16B) LDS loads.
__device__ inline v16h ld_frag2(const _Float16* p0, const _Float16* p1) {
    v8h a = *(const v8h*)p0;
    v8h b = *(const v8h*)p1;
    v16h r;
#pragma unroll
    for (int i = 0; i < 8; ++i) { r[i] = a[i]; r[i + 8] = b[i]; }
    return r;
}
// 16 contiguous halfs (16B-aligned) as two b128 loads.
__device__ inline v16h ld16(const _Float16* p) { return ld_frag2(p, p + 8); }

__global__ __launch_bounds__(256)
void flash_attn_f16_wmma(const float* __restrict__ Vg,   // "inputs" (values)
                         const float* __restrict__ Qg,   // "query"
                         const float* __restrict__ Kg,   // "key"
                         const int*   __restrict__ Mg,   // "mask" [B,1,S,S]
                         float* __restrict__ Og)         // out [B,H,S,D]
{
    __shared__ alignas(16) _Float16 sK[64 * KSTRIDE];        // 64 keys x 64 d (padded)
    __shared__ alignas(16) _Float16 sV[64 * KSTRIDE];        // 64 d x 64 keys (transposed, padded)
    __shared__ alignas(16) _Float16 sP[8 * 16 * KSTRIDE];    // per-wave 16x64 P (padded)

    const int tid  = threadIdx.x;
    const int w    = tid >> 5;          // wave id 0..7
    const int lane = tid & 31;
    const int l16  = lane & 15;
    const int hi   = (lane >> 4) & 1;   // 0: lanes 0-15, 1: lanes 16-31

    const int bh = blockIdx.y;                 // b*H + h
    const int bb = bh >> 3;                    // batch (H = 8)
    const int q0 = blockIdx.x * 128 + w * 16;  // first query row of this wave

    // ---- Load Q fragments (A-layout f16), pre-scaled by SCALE*log2e -------
    const int qrow = q0 + l16;
    const int d0   = hi ? 8 : 0;
    const float* qp = Qg + ((size_t)bh * ATT_S + qrow) * ATT_D;
    v16h Aq0, Aq1;
    {
        float qf[32];
#pragma unroll
        for (int j = 0; j < 2; ++j) {
            float4 f0 = *(const float4*)(qp + d0 + 32 * j);
            float4 f1 = *(const float4*)(qp + d0 + 32 * j + 4);
            float4 f2 = *(const float4*)(qp + d0 + 32 * j + 16);
            float4 f3 = *(const float4*)(qp + d0 + 32 * j + 20);
            qf[16*j+0]=f0.x;  qf[16*j+1]=f0.y;  qf[16*j+2]=f0.z;  qf[16*j+3]=f0.w;
            qf[16*j+4]=f1.x;  qf[16*j+5]=f1.y;  qf[16*j+6]=f1.z;  qf[16*j+7]=f1.w;
            qf[16*j+8]=f2.x;  qf[16*j+9]=f2.y;  qf[16*j+10]=f2.z; qf[16*j+11]=f2.w;
            qf[16*j+12]=f3.x; qf[16*j+13]=f3.y; qf[16*j+14]=f3.z; qf[16*j+15]=f3.w;
        }
#pragma unroll
        for (int i = 0; i < 16; ++i) {
            Aq0[i] = (_Float16)(qf[i]      * SCALE_L2E);
            Aq1[i] = (_Float16)(qf[i + 16] * SCALE_L2E);
        }
    }

    // Constant all-ones B fragment (virtual extra V column -> row sums).
    v16h ones;
#pragma unroll
    for (int i = 0; i < 16; ++i) ones[i] = (_Float16)1.0f;

    // ---- Flash accumulators: acc[0..3] = O numerator, acc[4] = denominator
    v8f acc[5];
#pragma unroll
    for (int c = 0; c < 5; ++c) acc[c] = (v8f)(0.0f);
    float mrow[8];
#pragma unroll
    for (int r = 0; r < 8; ++r) mrow[r] = -__builtin_inff();

    _Float16* myP = &sP[w * (16 * KSTRIDE)];

    // cooperative K/V tile load indexing: 64x64 f32, 16 floats/thread
    const int trow = tid >> 2;          // 0..63 (key row / v row)
    const int tcol = (tid & 3) * 16;    // 0,16,32,48

    const size_t kvbase = (size_t)bh * ATT_S * ATT_D;

    // ---- software pipeline: prefetch tile 0 into registers ---------------
    float4 pk[4], pv[4];
    {
        const float* kp = Kg + kvbase + (size_t)trow * ATT_D + tcol;
        const float* vp = Vg + kvbase + (size_t)trow * ATT_D + tcol;
#pragma unroll
        for (int j = 0; j < 4; ++j) {
            pk[j] = *(const float4*)(kp + 4 * j);
            pv[j] = *(const float4*)(vp + 4 * j);
        }
    }

    const int NT = ATT_S / 64;   // 32 tiles of 64 keys
    for (int kt = 0; kt < NT; ++kt) {
        const int k0 = kt * 64;

        // ---- registers -> LDS (f32->f16), V stored transposed -------------
        {
            _Float16* kd = &sK[trow * KSTRIDE + tcol];
            float kv[16] = {pk[0].x, pk[0].y, pk[0].z, pk[0].w,
                            pk[1].x, pk[1].y, pk[1].z, pk[1].w,
                            pk[2].x, pk[2].y, pk[2].z, pk[2].w,
                            pk[3].x, pk[3].y, pk[3].z, pk[3].w};
#pragma unroll
            for (int i = 0; i < 16; ++i) kd[i] = (_Float16)kv[i];
            float vv[16] = {pv[0].x, pv[0].y, pv[0].z, pv[0].w,
                            pv[1].x, pv[1].y, pv[1].z, pv[1].w,
                            pv[2].x, pv[2].y, pv[2].z, pv[2].w,
                            pv[3].x, pv[3].y, pv[3].z, pv[3].w};
#pragma unroll
            for (int i = 0; i < 16; ++i)
                sV[(tcol + i) * KSTRIDE + trow] = (_Float16)vv[i];
        }
        __syncthreads();

        // ---- prefetch next tile while computing this one ------------------
        if (kt + 1 < NT) {
            const float* kp = Kg + kvbase + (size_t)(k0 + 64 + trow) * ATT_D + tcol;
            const float* vp = Vg + kvbase + (size_t)(k0 + 64 + trow) * ATT_D + tcol;
#pragma unroll
            for (int j = 0; j < 4; ++j) {
                pk[j] = *(const float4*)(kp + 4 * j);
                pv[j] = *(const float4*)(vp + 4 * j);
            }
        }

        // ---- S(log2) = (Q*scale*log2e) K^T for four 16-key groups ---------
        v8f s[4];
#pragma unroll
        for (int g = 0; g < 4; ++g) {
            const int krow = g * 16 + l16;
            const int dofs = hi ? 8 : 0;
            v16h b0 = ld_frag2(&sK[krow * KSTRIDE + dofs], &sK[krow * KSTRIDE + dofs + 16]);        // d 0..31
            v16h b1 = ld_frag2(&sK[krow * KSTRIDE + 32 + dofs], &sK[krow * KSTRIDE + 48 + dofs]);   // d 32..63
            v8f cz = (v8f)(0.0f);
            cz = __builtin_amdgcn_wmma_f32_16x16x32_f16(false, Aq0, false, b0,
                                                        (short)0, cz, false, false);
            cz = __builtin_amdgcn_wmma_f32_16x16x32_f16(false, Aq1, false, b1,
                                                        (short)0, cz, false, false);
            s[g] = cz;
        }

        // ---- mask (broadcast over heads) ----------------------------------
#pragma unroll
        for (int g = 0; g < 4; ++g) {
#pragma unroll
            for (int r = 0; r < 8; ++r) {
                const int qq = q0 + r + hi * 8;
                const int kk = k0 + g * 16 + l16;
                const int mv = Mg[((size_t)bb * ATT_S + qq) * ATT_S + kk];
                s[g][r] = mv ? s[g][r] : NEG_L2E;
            }
        }

        // ---- online softmax (log2 domain); denominator via ones-WMMA ------
#pragma unroll
        for (int r = 0; r < 8; ++r) {
            float rm = fmaxf(fmaxf(s[0][r], s[1][r]), fmaxf(s[2][r], s[3][r]));
#pragma unroll
            for (int off = 8; off >= 1; off >>= 1)
                rm = fmaxf(rm, __shfl_xor(rm, off, 32));
            const float mnew  = fmaxf(mrow[r], rm);
            const float alpha = __builtin_amdgcn_exp2f(mrow[r] - mnew);
            mrow[r] = mnew;
#pragma unroll
            for (int g = 0; g < 4; ++g)
                s[g][r] = __builtin_amdgcn_exp2f(s[g][r] - mnew);
#pragma unroll
            for (int c = 0; c < 5; ++c) acc[c][r] *= alpha;
        }

        // ---- P (C-layout f32) -> LDS f16, reload in A-layout --------------
#pragma unroll
        for (int g = 0; g < 4; ++g) {
#pragma unroll
            for (int r = 0; r < 8; ++r) {
                const int row = r + hi * 8;
                const int col = g * 16 + l16;
                myP[row * KSTRIDE + col] = (_Float16)s[g][r];
            }
        }
        const int prow = l16;
        const int pcol = hi ? 8 : 0;
        v16h pa0 = ld_frag2(&myP[prow * KSTRIDE + pcol],
                            &myP[prow * KSTRIDE + pcol + 16]);   // keys 0..31
        v16h pa1 = ld_frag2(&myP[prow * KSTRIDE + pcol + 32],
                            &myP[prow * KSTRIDE + pcol + 48]);   // keys 32..63

        // ---- acc[0..3] += P V ; acc[4] += P * ones (row sums) -------------
#pragma unroll
        for (int c = 0; c < 4; ++c) {
            const int dd   = c * 16 + l16;
            const int kofs = hi ? 8 : 0;
            v16h vb0 = ld_frag2(&sV[dd * KSTRIDE + kofs], &sV[dd * KSTRIDE + kofs + 16]);        // keys 0..31
            v16h vb1 = ld_frag2(&sV[dd * KSTRIDE + 32 + kofs], &sV[dd * KSTRIDE + 48 + kofs]);   // keys 32..63
            acc[c] = __builtin_amdgcn_wmma_f32_16x16x32_f16(false, pa0, false, vb0,
                                                            (short)0, acc[c], false, false);
            acc[c] = __builtin_amdgcn_wmma_f32_16x16x32_f16(false, pa1, false, vb1,
                                                            (short)0, acc[c], false, false);
        }
        acc[4] = __builtin_amdgcn_wmma_f32_16x16x32_f16(false, pa0, false, ones,
                                                        (short)0, acc[4], false, false);
        acc[4] = __builtin_amdgcn_wmma_f32_16x16x32_f16(false, pa1, false, ones,
                                                        (short)0, acc[4], false, false);
        __syncthreads();
    }

    // ---- epilogue: normalize and store -----------------------------------
#pragma unroll
    for (int r = 0; r < 8; ++r) {
        const float inv = 1.0f / acc[4][r];
        const int qq = q0 + r + hi * 8;
#pragma unroll
        for (int c = 0; c < 4; ++c) {
            const int dd = c * 16 + l16;
            Og[((size_t)bh * ATT_S + qq) * ATT_D + dd] = acc[c][r] * inv;
        }
    }
}

extern "C" void kernel_launch(void* const* d_in, const int* in_sizes, int n_in,
                              void* d_out, int out_size, void* d_ws, size_t ws_size,
                              hipStream_t stream) {
    (void)in_sizes; (void)n_in; (void)out_size; (void)d_ws; (void)ws_size;
    const float* Vg = (const float*)d_in[0];   // "inputs" (values)
    const float* Qg = (const float*)d_in[1];   // "query"
    const float* Kg = (const float*)d_in[2];   // "key"
    const int*   Mg = (const int*)d_in[3];     // "mask"
    float* Og = (float*)d_out;

    dim3 grid(ATT_S / 128, ATT_B * ATT_H);     // (16, 32)
    dim3 block(256);
    flash_attn_f16_wmma<<<grid, block, 0, stream>>>(Vg, Qg, Kg, Mg, Og);
}
